// Block_37632503448156
// MI455X (gfx1250) — matrix operand order repaired and checked by reading, not compiled, not run on hardware
//
#include <hip/hip_runtime.h>
#include <hip/hip_bf16.h>

typedef unsigned short u16;
typedef unsigned int u32;
typedef unsigned long long u64;
typedef __attribute__((ext_vector_type(16))) __bf16 v16bf;
typedef __attribute__((ext_vector_type(8)))  float  v8f;
typedef __attribute__((ext_vector_type(4)))  unsigned v4u;
typedef __attribute__((ext_vector_type(8)))  int v8i;
typedef __attribute__((ext_vector_type(4)))  int v4i;

// ---------------- constants (from reference) ----------------
#define BB 2
#define TT 2048
#define CC 1024
#define HH 16
#define HDD 64
#define FFD 4096
#define MM (BB * TT)        // 4096 rows
#define S3 (3 * CC)         // 3072

// ---------------- TDM availability ----------------
#ifdef __has_builtin
#if __has_builtin(__builtin_amdgcn_tensor_load_to_lds)
#define HAVE_TDM 1
#endif
#endif
#ifndef HAVE_TDM
#define HAVE_TDM 0
#endif

// ---------------- helpers ----------------
__device__ __forceinline__ u16 f2bf(float f) {
    u32 u = __builtin_bit_cast(u32, f);
    return (u16)((u + 0x7FFFu + ((u >> 16) & 1u)) >> 16);
}

// load a 16-element bf16 fragment: 8 elems at p, 8 elems at p+gap
__device__ __forceinline__ v16bf frag_ld2(const u16* p, int gap) {
    union { uint4 q[2]; v16bf v; } u;
    u.q[0] = *(const uint4*)(p);
    u.q[1] = *(const uint4*)(p + gap);
    return u.v;
}

__device__ __forceinline__ v8f wmma_bf16(v16bf a, v16bf b, v8f c) {
    return __builtin_amdgcn_wmma_f32_16x16x32_bf16(
        false, a, false, b, (short)0, c, false, false);
}

__device__ __forceinline__ void tensor_wait0() {
#if HAVE_TDM
#if __has_builtin(__builtin_amdgcn_s_wait_tensorcnt)
    __builtin_amdgcn_s_wait_tensorcnt(0);
#else
    asm volatile("s_wait_tensorcnt 0x0" ::: "memory");
#endif
#endif
}

#if HAVE_TDM
// TDM 2D tile load: 128 rows x 64 bf16, global row stride = K elements,
// LDS padded to 144B rows via pad_enable (4 DWORDs after every 32 DWORDs).
// D# layout per CDNA5 ISA 8.3 (group0) / 8.4 (group1).
__device__ __forceinline__ void tdm_load_tile(const u16* gptr, u32 lds_addr, int K) {
    u64 ga = (u64)(size_t)gptr;
    v4u g0;
    g0.x = 1u;                                   // count=1, user descriptor
    g0.y = lds_addr;                             // lds_addr [63:32]
    g0.z = (u32)ga;                              // global_addr low
    g0.w = (u32)(ga >> 32) | (2u << 30);         // global_addr high | type=2
    v8i g1;
    // data_size=1 (2B) [17:16]; pad_enable [20]; pad_interval=4 (32 DW) [24:22];
    // pad_amount=3 (4 DW) [31:25]; workgroup_mask=0
    g1[0] = (1 << 16) | (1 << 20) | (4 << 22) | (3 << 25);
    g1[1] = (int)((u32)K << 16);                 // tensor_dim0[15:0] -> [63:48]
    g1[2] = (int)(((u32)K >> 16) | (128u << 16));// tensor_dim0[31:16], tensor_dim1=128 lo
    g1[3] = (int)(64u << 16);                    // tensor_dim1 hi=0, tile_dim0=64
    g1[4] = 128;                                 // tile_dim1=128, tile_dim2=0
    g1[5] = K;                                   // tensor_dim0_stride low32
    g1[6] = 0;                                   // stride hi16, dim1_stride lo16
    g1[7] = 0;
    v4i zz = {0, 0, 0, 0};
#if __clang_major__ >= 23
    v8i z8 = {0, 0, 0, 0, 0, 0, 0, 0};
    __builtin_amdgcn_tensor_load_to_lds(g0, g1, zz, zz, z8, 0);
#else
    __builtin_amdgcn_tensor_load_to_lds(g0, g1, zz, zz, 0);
#endif
}
#endif

// ---------------- LayerNorm: f32 [M][C] -> bf16 [M][C] ----------------
__global__ __launch_bounds__(256) void ln_kernel(const float* __restrict__ x,
                                                 const float* __restrict__ g,
                                                 const float* __restrict__ be,
                                                 u16* __restrict__ out) {
    const int row = blockIdx.x;
    const int tid = threadIdx.x;
    const float* xr = x + (size_t)row * CC;
    float v[4], s1 = 0.f, s2 = 0.f;
    for (int i = 0; i < 4; ++i) {
        v[i] = xr[tid + 256 * i];
        s1 += v[i];
        s2 += v[i] * v[i];
    }
    __shared__ float r1[256], r2[256];
    r1[tid] = s1; r2[tid] = s2;
    __syncthreads();
    for (int off = 128; off > 0; off >>= 1) {
        if (tid < off) { r1[tid] += r1[tid + off]; r2[tid] += r2[tid + off]; }
        __syncthreads();
    }
    const float mean = r1[0] * (1.f / CC);
    const float var  = r2[0] * (1.f / CC) - mean * mean;
    const float rstd = rsqrtf(var + 1e-5f);
    u16* orow = out + (size_t)row * CC;
    for (int i = 0; i < 4; ++i) {
        int c = tid + 256 * i;
        orow[c] = f2bf((v[i] - mean) * rstd * g[c] + be[c]);
    }
}

// ---------------- weight packing ----------------
// fuse Wq/Wk/Wv [H,C,HD] f32 -> [3C][C] bf16, n-major (k contiguous)
__global__ __launch_bounds__(256) void pack_qkvT(const float* __restrict__ Wq,
                                                 const float* __restrict__ Wk,
                                                 const float* __restrict__ Wv,
                                                 u16* __restrict__ out) {
    int idx = blockIdx.x * 256 + threadIdx.x;        // over 3C*C
    int n = idx >> 10, k = idx & (CC - 1);
    int sel = n >> 10, hd = n & (CC - 1);
    int head = hd >> 6, d = hd & 63;
    const float* W = (sel == 0) ? Wq : ((sel == 1) ? Wk : Wv);
    out[idx] = f2bf(W[(size_t)head * CC * HDD + (size_t)k * HDD + d]);
}

// transpose f32 [K][N] -> bf16 [N][K]
__global__ __launch_bounds__(256) void packT(const float* __restrict__ W,
                                             u16* __restrict__ out, int K, int N) {
    int idx = blockIdx.x * 256 + threadIdx.x;        // over N*K
    int n = idx / K, k = idx - n * K;
    out[idx] = f2bf(W[(size_t)k * N + n]);
}

// ---------------- bf16 WMMA GEMM: out = A[M,K] x Bt[N,K]^T ----------------
// 128x128 tile, BK=64, double-buffered LDS; tiles fetched by the Tensor Data
// Mover (TENSORcnt) when available, else vectorized b128 loads.
// epilogue: +bias, relu, +residual, store f32 and/or bf16
__global__ __launch_bounds__(256) void gemm_bf16(const u16* __restrict__ A,
                                                 const u16* __restrict__ Bt,
                                                 const float* __restrict__ bias,
                                                 const float* __restrict__ resid,
                                                 float* __restrict__ outf,
                                                 u16* __restrict__ outbf,
                                                 int M, int N, int K, int relu) {
    __shared__ __align__(16) u16 sA[2][128][72];   // 144B row stride
    __shared__ __align__(16) u16 sB[2][128][72];
    const int tid  = threadIdx.x;
    const int lane = tid & 31, wid = tid >> 5;
    const int wm = wid & 3, wn = wid >> 2;
    const int m0 = blockIdx.y * 128, n0 = blockIdx.x * 128;
    const int lrow = tid >> 1, lseg = (tid & 1) * 32;
    (void)lrow; (void)lseg;

    const v8f vzero = {0.f, 0.f, 0.f, 0.f, 0.f, 0.f, 0.f, 0.f};
    v8f acc[2][4];
    for (int mt = 0; mt < 2; ++mt)
        for (int nt = 0; nt < 4; ++nt) acc[mt][nt] = vzero;

    auto load_tiles = [&](int buf, int k0) {
#if HAVE_TDM
        if (wid == 0) {
            tdm_load_tile(A  + (size_t)m0 * K + k0, (u32)(size_t)&sA[buf][0][0], K);
            tdm_load_tile(Bt + (size_t)n0 * K + k0, (u32)(size_t)&sB[buf][0][0], K);
        }
#else
        const u16* ga = A  + (size_t)(m0 + lrow) * K + k0 + lseg;
        const u16* gb = Bt + (size_t)(n0 + lrow) * K + k0 + lseg;
        for (int i = 0; i < 4; ++i) {
            *(uint4*)&sA[buf][lrow][lseg + 8 * i] = *(const uint4*)(ga + 8 * i);
            *(uint4*)&sB[buf][lrow][lseg + 8 * i] = *(const uint4*)(gb + 8 * i);
        }
#endif
    };
    auto wait_tiles = [&]() {
#if HAVE_TDM
        if (wid == 0) tensor_wait0();
#endif
    };

    const int nk = K >> 6;
    load_tiles(0, 0);
    wait_tiles();
    __syncthreads();

    const int kba = (lane >> 4) * 8;
    const int kbb = (lane >> 4) * 8;
    for (int kt = 0; kt < nk; ++kt) {
        const int cur = kt & 1;
        if (kt + 1 < nk) load_tiles(cur ^ 1, (kt + 1) << 6);

        for (int ks = 0; ks < 2; ++ks) {
            v16bf afrag[2], bfrag[4];
            for (int mt = 0; mt < 2; ++mt)
                afrag[mt] = frag_ld2(&sA[cur][32 * wm + 16 * mt + (lane & 15)][ks * 32 + kba], 16);
            for (int nt = 0; nt < 4; ++nt)
                bfrag[nt] = frag_ld2(&sB[cur][64 * wn + 16 * nt + (lane & 15)][ks * 32 + kbb], 16);
            for (int mt = 0; mt < 2; ++mt)
                for (int nt = 0; nt < 4; ++nt)
                    acc[mt][nt] = wmma_bf16(afrag[mt], bfrag[nt], acc[mt][nt]);
        }
        wait_tiles();
        __syncthreads();
    }

    for (int mt = 0; mt < 2; ++mt) {
        for (int nt = 0; nt < 4; ++nt) {
            for (int r = 0; r < 8; ++r) {
                int grow = m0 + 32 * wm + 16 * mt + r + 8 * (lane >> 4);
                int gcol = n0 + 64 * wn + 16 * nt + (lane & 15);
                float v = acc[mt][nt][r];
                if (bias)  v += bias[gcol];
                if (relu)  v = fmaxf(v, 0.f);
                if (resid) v += resid[(size_t)grow * N + gcol];
                if (outf)  outf[(size_t)grow * N + gcol] = v;
                if (outbf) outbf[(size_t)grow * N + gcol] = f2bf(v);
            }
        }
    }
}

// ---------------- flash attention (causal, 64-row Q tiles) ----------------
// qkv: bf16 [B*T][3C]; out: bf16 [B*T][C]
__global__ __launch_bounds__(256) void attn_kernel(const u16* __restrict__ qkv,
                                                   u16* __restrict__ attn_out) {
    __shared__ __align__(16) u16 sQ[64][72];
    __shared__ __align__(16) u16 sK[64][72];
    __shared__ __align__(16) u16 sVt[64][72];   // [d][j]
    __shared__ __align__(16) u16 sP[64][72];
    __shared__ __align__(16) float sS[64][72];
    __shared__ float sM[64], sL[64], sF[64];

    const int tid  = threadIdx.x;
    const int lane = tid & 31, wid = tid >> 5;
    const int wm = wid & 3, wn = wid >> 2;     // wave: 16 rows x 32 cols
    const int qb = blockIdx.x, h = blockIdx.y, b = blockIdx.z;
    const size_t rowbase = (size_t)b * TT;

    // load Q tile (64x64 bf16)
    const int lr = tid >> 2, ls = (tid & 3) * 16;
    {
        const u16* gq = qkv + (rowbase + qb * 64 + lr) * S3 + h * HDD + ls;
        *(uint4*)&sQ[lr][ls]     = *(const uint4*)(gq);
        *(uint4*)&sQ[lr][ls + 8] = *(const uint4*)(gq + 8);
    }
    if (tid < 64) { sM[tid] = -1e30f; sL[tid] = 0.f; }

    const v8f vzero = {0.f, 0.f, 0.f, 0.f, 0.f, 0.f, 0.f, 0.f};
    v8f o[2] = {vzero, vzero};

    for (int kb = 0; kb <= qb; ++kb) {
        __syncthreads();
        // load K tile and V tile (V transposed into LDS)
        const u16* gk = qkv + (rowbase + kb * 64 + lr) * S3 + CC + h * HDD + ls;
        *(uint4*)&sK[lr][ls]     = *(const uint4*)(gk);
        *(uint4*)&sK[lr][ls + 8] = *(const uint4*)(gk + 8);
        const u16* gv = qkv + (rowbase + kb * 64 + lr) * S3 + 2 * CC + h * HDD + ls;
        u16 tmp[16];
        *(uint4*)&tmp[0] = *(const uint4*)(gv);
        *(uint4*)&tmp[8] = *(const uint4*)(gv + 8);
        for (int i = 0; i < 16; ++i) sVt[ls + i][lr] = tmp[i];
        __syncthreads();

        // S = Q @ K^T (scaled), masked
        v8f s_acc[2] = {vzero, vzero};
        for (int ks = 0; ks < 2; ++ks) {
            v16bf a = frag_ld2(&sQ[16 * wm + (lane & 15)][ks * 32 + ((lane >> 4) * 8)], 16);
            for (int nt = 0; nt < 2; ++nt) {
                v16bf bb = frag_ld2(&sK[32 * wn + 16 * nt + (lane & 15)][ks * 32 + ((lane >> 4) * 8)], 16);
                s_acc[nt] = wmma_bf16(a, bb, s_acc[nt]);
            }
        }
        for (int nt = 0; nt < 2; ++nt) {
            for (int r = 0; r < 8; ++r) {
                int rl = 16 * wm + r + 8 * (lane >> 4);
                int cl = 32 * wn + 16 * nt + (lane & 15);
                int gqi = qb * 64 + rl, gki = kb * 64 + cl;
                float v = s_acc[nt][r] * 0.125f;       // 1/sqrt(64)
                sS[rl][cl] = (gki <= gqi) ? v : -1e30f;
            }
        }
        __syncthreads();

        // online softmax per row (threads 0..63)
        if (tid < 64) {
            const int row = tid;
            float mprev = sM[row];
            float rmax = mprev;
            for (int j = 0; j < 64; ++j) rmax = fmaxf(rmax, sS[row][j]);
            float f = __expf(mprev - rmax);
            float sum = 0.f;
            for (int j = 0; j < 64; ++j) {
                float p = __expf(sS[row][j] - rmax);
                sP[row][j] = f2bf(p);
                sum += p;
            }
            sL[row] = sL[row] * f + sum;
            sM[row] = rmax;
            sF[row] = f;
        }
        __syncthreads();

        // rescale O, then O += P @ V
        for (int r = 0; r < 8; ++r) {
            int rl = 16 * wm + r + 8 * (lane >> 4);
            float f = sF[rl];
            o[0][r] *= f;
            o[1][r] *= f;
        }
        for (int ks = 0; ks < 2; ++ks) {
            v16bf a = frag_ld2(&sP[16 * wm + (lane & 15)][ks * 32 + ((lane >> 4) * 8)], 16);
            for (int nt = 0; nt < 2; ++nt) {
                v16bf bb = frag_ld2(&sVt[32 * wn + 16 * nt + (lane & 15)][ks * 32 + ((lane >> 4) * 8)], 16);
                o[nt] = wmma_bf16(a, bb, o[nt]);
            }
        }
    }

    // write normalized output (bf16, head-concat layout)
    for (int nt = 0; nt < 2; ++nt) {
        for (int r = 0; r < 8; ++r) {
            int rl = 16 * wm + r + 8 * (lane >> 4);
            int cl = 32 * wn + 16 * nt + (lane & 15);
            float v = o[nt][r] / sL[rl];
            attn_out[(rowbase + qb * 64 + rl) * CC + h * HDD + cl] = f2bf(v);
        }
    }
}

// ---------------- host orchestration ----------------
extern "C" void kernel_launch(void* const* d_in, const int* in_sizes, int n_in,
                              void* d_out, int out_size, void* d_ws, size_t ws_size,
                              hipStream_t stream) {
    (void)in_sizes; (void)n_in; (void)out_size; (void)ws_size;
    const float* x   = (const float*)d_in[0];
    const float* Wq  = (const float*)d_in[1];
    const float* Wk  = (const float*)d_in[2];
    const float* Wv  = (const float*)d_in[3];
    const float* Wo  = (const float*)d_in[4];
    const float* bo  = (const float*)d_in[5];
    const float* W1  = (const float*)d_in[6];
    const float* b1  = (const float*)d_in[7];
    const float* W2  = (const float*)d_in[8];
    const float* b2  = (const float*)d_in[9];
    const float* g1  = (const float*)d_in[10];
    const float* be1 = (const float*)d_in[11];
    const float* g2  = (const float*)d_in[12];
    const float* be2 = (const float*)d_in[13];
    float* out = (float*)d_out;

    // workspace carve-up (256B aligned)
    char* w = (char*)d_ws;
    auto alloc = [&](size_t bytes) {
        char* p = w;
        w += (bytes + 255) & ~(size_t)255;
        return p;
    };
    u16*   wqkvT   = (u16*)alloc((size_t)S3 * CC * 2);      // 6 MB
    u16*   woT     = (u16*)alloc((size_t)CC * CC * 2);      // 2 MB
    u16*   w1T     = (u16*)alloc((size_t)FFD * CC * 2);     // 8 MB
    u16*   w2T     = (u16*)alloc((size_t)CC * FFD * 2);     // 8 MB
    u16*   h_bf    = (u16*)alloc((size_t)MM * CC * 2);      // 8 MB (reused as h2)
    u16*   qkv_bf  = (u16*)alloc((size_t)MM * S3 * 2);      // 24 MB
    u16*   attn_bf = (u16*)alloc((size_t)MM * CC * 2);      // 8 MB (contiguous after qkv)
    float* x1      = (float*)alloc((size_t)MM * CC * 4);    // 16 MB
    u16*   ff_bf   = qkv_bf;   // 32 MB alias over dead qkv_bf+attn_bf
    u16*   h2_bf   = h_bf;     // alias; h dead after QKV GEMM

    // 1) pack weights to bf16 [N][K]
    pack_qkvT<<<(S3 * CC) / 256, 256, 0, stream>>>(Wq, Wk, Wv, wqkvT);
    packT<<<(CC * CC) / 256, 256, 0, stream>>>(Wo, woT, CC, CC);
    packT<<<(FFD * CC) / 256, 256, 0, stream>>>(W1, w1T, CC, FFD);
    packT<<<(CC * FFD) / 256, 256, 0, stream>>>(W2, w2T, FFD, CC);

    // 2) pre-LN 1
    ln_kernel<<<MM, 256, 0, stream>>>(x, g1, be1, h_bf);

    // 3) QKV projection: [M,C] x [C,3C] -> bf16 [M,3C]
    gemm_bf16<<<dim3(S3 / 128, MM / 128), 256, 0, stream>>>(
        h_bf, wqkvT, nullptr, nullptr, nullptr, qkv_bf, MM, S3, CC, 0);

    // 4) causal flash attention -> bf16 [M,C]
    attn_kernel<<<dim3(TT / 64, HH, BB), 256, 0, stream>>>(qkv_bf, attn_bf);

    // 5) output projection + bias + residual(x) -> f32 x1
    gemm_bf16<<<dim3(CC / 128, MM / 128), 256, 0, stream>>>(
        attn_bf, woT, bo, x, x1, nullptr, MM, CC, CC, 0);

    // 6) pre-LN 2
    ln_kernel<<<MM, 256, 0, stream>>>(x1, g2, be2, h2_bf);

    // 7) FFN up + bias + ReLU -> bf16 [M,FF]
    gemm_bf16<<<dim3(FFD / 128, MM / 128), 256, 0, stream>>>(
        h2_bf, w1T, b1, nullptr, nullptr, ff_bf, MM, FFD, CC, 1);

    // 8) FFN down + bias + residual(x1) -> f32 out
    gemm_bf16<<<dim3(CC / 128, MM / 128), 256, 0, stream>>>(
        ff_bf, w2T, b2, x1, out, nullptr, MM, CC, FFD, 0);
}
